// MultiHeadSelfAttention_18803366821902
// MI455X (gfx1250) — matrix operand order repaired and checked by reading, not compiled, not run on hardware
//
#include <hip/hip_runtime.h>
#include <hip/hip_bf16.h>

// Multi-head self-attention, B=4 S=2048 D=1024 H=16 DH=64, bf16 WMMA pipeline.
#define B_  4
#define S_  2048
#define D_  1024
#define H_  16
#define DH_ 64

typedef __attribute__((ext_vector_type(8)))  float        v8f;
typedef __attribute__((ext_vector_type(8)))  __bf16       v8bf;
typedef __attribute__((ext_vector_type(16))) __bf16       v16bf;
typedef __attribute__((ext_vector_type(4)))  unsigned int v4u;
typedef __attribute__((ext_vector_type(8)))  int          v8i;
typedef __attribute__((ext_vector_type(4)))  int          v4i;

static __device__ __forceinline__ v16bf bf_combine(v8bf lo, v8bf hi) {
  v16bf r;
#pragma unroll
  for (int i = 0; i < 8; i++) { r[i] = lo[i]; r[i + 8] = hi[i]; }
  return r;
}

static __device__ __forceinline__ v8bf cvt8(float4 a, float4 b) {
  v8bf r;
  r[0] = (__bf16)a.x; r[1] = (__bf16)a.y; r[2] = (__bf16)a.z; r[3] = (__bf16)a.w;
  r[4] = (__bf16)b.x; r[5] = (__bf16)b.y; r[6] = (__bf16)b.z; r[7] = (__bf16)b.w;
  return r;
}

static __device__ __forceinline__ v8f wmma_bf16(v16bf a, v16bf b, v8f c) {
  // D(16x16,f32) = A(16x32,bf16) * B(32x16,bf16) + C
  return __builtin_amdgcn_wmma_f32_16x16x32_bf16(false, a, false, b, (short)0, c,
                                                 false, false);
}

// ---------------------------------------------------------------------------
// TDM: stage a 32x64 bf16 tile (row stride 64 elements in memory) into LDS,
// inserting 4 pad dwords after every 32 data dwords (-> 144B LDS row stride,
// conflict-free B128 fragment reads).  D# built per CDNA5 ISA ch.8.
// 6-arg builtin form (clang-23 / therock-10.0 headers).
// ---------------------------------------------------------------------------
static __device__ __forceinline__ void tdm_load_k_tile(const __bf16* gsrc,
                                                       unsigned lds_off) {
  unsigned long long ga = (unsigned long long)(size_t)gsrc;
  v4u g0;
  g0[0] = 1u;                                   // count=1, user descriptor
  g0[1] = lds_off;                              // lds_addr (bytes)
  g0[2] = (unsigned)(ga & 0xffffffffu);         // global_addr[31:0]
  g0[3] = (unsigned)((ga >> 32) & 0x01ffffffu)  // global_addr[56:32]
          | 0x80000000u;                        // type=2 ("image")
  v8i g1;
  g1[0] = (int)((1u << 16)      // data_size = 2 bytes
              | (1u << 20)      // pad_enable
              | (4u << 22)      // pad_interval: 32 dwords
              | (3u << 25));    // pad_amount: 4 dwords
  g1[1] = (int)(64u << 16);     // tensor_dim0 = 64 (low 16 bits, bits 63:48)
  g1[2] = 0;                    // tensor_dim0 hi / tensor_dim1 lo
  g1[3] = (int)(16u             // tensor_dim1 hi (tensor_dim1 = 1<<20, no OOB)
              | (64u << 16));   // tile_dim0 = 64
  g1[4] = 32;                   // tile_dim1 = 32, tile_dim2 = 0
  g1[5] = 64;                   // tensor_dim0_stride = 64
  g1[6] = 0;
  g1[7] = 0;
  v4i z4 = {0, 0, 0, 0};
  v8i z8 = {0, 0, 0, 0, 0, 0, 0, 0};
  __builtin_amdgcn_tensor_load_to_lds(g0, g1, z4, z4, z8, 0);
}

// ---------------------------------------------------------------------------
// Kernel 1: convert the 4 weight matrices (in,out) fp32 -> bf16 transposed
// (out,in), so GEMM B-fragments are two k-contiguous 16B global loads per lane.
// ---------------------------------------------------------------------------
__global__ void wconv_kernel(const float* __restrict__ Wq, const float* __restrict__ Wk,
                             const float* __restrict__ Wv, const float* __restrict__ Wo,
                             __bf16* __restrict__ WtAll) {
  const int which = blockIdx.y;
  const float* src = (which == 0) ? Wq : (which == 1) ? Wk : (which == 2) ? Wv : Wo;
  __bf16* dst = WtAll + (size_t)which * D_ * D_;
  int idx = blockIdx.x * blockDim.x + threadIdx.x;  // coalesced read
  int k = idx / D_, n = idx % D_;
  dst[(size_t)n * D_ + k] = (__bf16)src[idx];
}

// ---------------------------------------------------------------------------
// Kernel 2: fused QKV projection.  X(8192x1024,f32) @ W(1024x1024) + b,
// Q additionally scaled by 1/sqrt(DH).  Output bf16 in (B,H,S,DH) layout.
// Block = 256 threads = 8 waves, wave grid 4(M) x 2(N), wave tile 32x64
// (2 M-tiles x 4 N-tiles = 8 WMMAs per K-step, 2 B128 loads per WMMA).
// ---------------------------------------------------------------------------
__global__ void __launch_bounds__(256)
qkv_proj_kernel(const float* __restrict__ Xq, const float* __restrict__ Xk,
                const float* __restrict__ Xv, const __bf16* __restrict__ WtAll,
                const float* __restrict__ bq, const float* __restrict__ bk,
                const float* __restrict__ bv, __bf16* __restrict__ outq,
                __bf16* __restrict__ outk, __bf16* __restrict__ outv) {
  const int z = blockIdx.z;
  const float* X = (z == 0) ? Xq : (z == 1) ? Xk : Xv;
  const __bf16* Wt = WtAll + (size_t)z * D_ * D_;
  const float* bias = (z == 0) ? bq : (z == 1) ? bk : bv;
  __bf16* out = (z == 0) ? outq : (z == 1) ? outk : outv;
  const float scale = (z == 0) ? 0.125f : 1.0f;  // 1/sqrt(64)

  const int lane = threadIdx.x & 31;
  const int wave = threadIdx.x >> 5;
  const int hf = lane >> 4;   // half-wave select
  const int lh = lane & 15;

  const int wm = wave & 3, wn = wave >> 2;
  const int mbase = blockIdx.x * 128 + wm * 32;
  const int nbase = blockIdx.y * 128 + wn * 64;

  v8f acc[2][4];
#pragma unroll
  for (int mt = 0; mt < 2; mt++)
#pragma unroll
    for (int nt = 0; nt < 4; nt++)
#pragma unroll
      for (int i = 0; i < 8; i++) acc[mt][nt][i] = 0.0f;

  const int rowA0 = mbase + lh;
  const int rowA1 = mbase + 16 + lh;

  for (int k0 = 0; k0 < D_; k0 += 32) {
    // A fragments (fp32 -> bf16 convert in VALU).
    // Lane layout: lane<16 row=lh K=[0..7|16..23]; lane>=16 K=[8..15|24..31]
    v16bf A0, A1;
    {
      const float* p = X + (size_t)rowA0 * D_ + k0 + 8 * hf;
      A0 = bf_combine(cvt8(*(const float4*)p, *(const float4*)(p + 4)),
                      cvt8(*(const float4*)(p + 16), *(const float4*)(p + 20)));
      const float* q = X + (size_t)rowA1 * D_ + k0 + 8 * hf;
      A1 = bf_combine(cvt8(*(const float4*)q, *(const float4*)(q + 4)),
                      cvt8(*(const float4*)(q + 16), *(const float4*)(q + 20)));
    }
#pragma unroll
    for (int nt = 0; nt < 4; nt++) {
      const __bf16* p = Wt + (size_t)(nbase + nt * 16 + lh) * D_ + k0 + 8 * hf;
      v16bf Bf = bf_combine(*(const v8bf*)p, *(const v8bf*)(p + 16));
      acc[0][nt] = wmma_bf16(A0, Bf, acc[0][nt]);
      acc[1][nt] = wmma_bf16(A1, Bf, acc[1][nt]);
    }
  }

  // Epilogue: bias, scale, scatter to (B,H,S,DH) bf16.
#pragma unroll
  for (int nt = 0; nt < 4; nt++) {
    const int col = nbase + nt * 16 + lh;
    const float bcol = bias[col];
    const int h = col >> 6, f = col & 63;
#pragma unroll
    for (int mt = 0; mt < 2; mt++) {
#pragma unroll
      for (int r = 0; r < 8; r++) {
        const int row = mbase + mt * 16 + r + 8 * hf;  // 0..8191
        const int b = row >> 11, s = row & 2047;
        float v = (acc[mt][nt][r] + bcol) * scale;
        out[((size_t)(b * H_ + h) * S_ + s) * DH_ + f] = (__bf16)v;
      }
    }
  }
}

// ---------------------------------------------------------------------------
// Kernel 3: flash attention over one (b,h), 128 query rows per block,
// online softmax, KV blocks of 32.  8 waves x 16 query rows each.
// K tile staged by the Tensor Data Mover (padded rows, conflict-free reads);
// V tile transposed into LDS by VALU; P relayout via per-wave LDS scratch.
// ---------------------------------------------------------------------------
#define KSTRIDE 72   // 64 data + 8 pad bf16 per K row (144 B, from TDM padding)
#define VSTRIDE 40   // padded V^T / P strides (80 B) -> 16 distinct banks
__global__ void __launch_bounds__(256)
flash_attn_kernel(const __bf16* __restrict__ qp, const __bf16* __restrict__ kp,
                  const __bf16* __restrict__ vp, const float* __restrict__ mask,
                  __bf16* __restrict__ ctx) {
  __shared__ __bf16 Klds[32 * KSTRIDE];      // K tile [key][feat], TDM-padded
  __shared__ __bf16 Vt[DH_ * VSTRIDE];       // V^T tile [feat][key]
  __shared__ __bf16 Pscr[8][16 * VSTRIDE];   // per-wave P scratch [row][key]

  const int bh = blockIdx.y;
  const int b = bh >> 4, h = bh & 15;
  const int lane = threadIdx.x & 31;
  const int wave = threadIdx.x >> 5;
  const int hf = lane >> 4, lh = lane & 15;
  const int qrow0 = blockIdx.x * 128 + wave * 16;

  const size_t headoff = (size_t)bh * S_ * DH_;
  const __bf16* Qh = qp + headoff;
  const __bf16* Kh = kp + headoff;
  const __bf16* Vh = vp + headoff;
  const unsigned klds_off = (unsigned)(size_t)(&Klds[0]);

  // Q A-fragments live in registers across the whole KV loop.
  v16bf Qf[2];
#pragma unroll
  for (int ks = 0; ks < 2; ks++) {
    const __bf16* p = Qh + (size_t)(qrow0 + lh) * DH_ + ks * 32 + 8 * hf;
    Qf[ks] = bf_combine(*(const v8bf*)p, *(const v8bf*)(p + 16));
  }

  float mrow[8], lrow[8];
  v8f O[4];
#pragma unroll
  for (int r = 0; r < 8; r++) { mrow[r] = -1e30f; lrow[r] = 0.0f; }
#pragma unroll
  for (int t = 0; t < 4; t++)
#pragma unroll
    for (int i = 0; i < 8; i++) O[t][i] = 0.0f;

  for (int kb = 0; kb < S_ / 32; kb++) {
    const int key0 = kb * 32;

    __syncthreads();  // previous iteration's LDS reads are done

    // One wave kicks the TDM copy of the K tile (EXEC-independent DMA).
    if (wave == 0) tdm_load_k_tile(Kh + (size_t)key0 * DH_, klds_off);

    // Meanwhile all threads stage V (32 keys x 64 feats) transposed into LDS.
    {
      const int keyy = threadIdx.x >> 3;
      const int feat0 = (threadIdx.x & 7) * 8;
      v8bf vv = *(const v8bf*)(Vh + (size_t)(key0 + keyy) * DH_ + feat0);
#pragma unroll
      for (int j = 0; j < 8; j++) Vt[(feat0 + j) * VSTRIDE + keyy] = vv[j];
    }

    // Prefetch next V block while we compute on this one.
    if (wave == 1 && key0 + 32 < S_)
      __builtin_prefetch(Vh + (size_t)(key0 + 32 + lane) * DH_, 0, 0);

    if (wave == 0) __builtin_amdgcn_s_wait_tensorcnt(0);
    __syncthreads();

    // Scores: S(16x32) = Q(16x64) @ K^T(64x32); two 16x16 C tiles.
    v8f Sc[2];
#pragma unroll
    for (int nt = 0; nt < 2; nt++) {
#pragma unroll
      for (int i = 0; i < 8; i++) Sc[nt][i] = 0.0f;
#pragma unroll
      for (int ks = 0; ks < 2; ks++) {
        const __bf16* p = Klds + (nt * 16 + lh) * KSTRIDE + ks * 32 + 8 * hf;
        v16bf Bf = bf_combine(*(const v8bf*)p, *(const v8bf*)(p + 16));
        Sc[nt] = wmma_bf16(Qf[ks], Bf, Sc[nt]);
      }
    }

    // Additive mask (per key column).
    const float mk0 = mask[b * S_ + key0 + lh];
    const float mk1 = mask[b * S_ + key0 + 16 + lh];
#pragma unroll
    for (int r = 0; r < 8; r++) { Sc[0][r] += mk0; Sc[1][r] += mk1; }

    // Online softmax (rows live in element r across the 16 lanes of a half).
#pragma unroll
    for (int r = 0; r < 8; r++) {
      float mx = fmaxf(Sc[0][r], Sc[1][r]);
      mx = fmaxf(mx, __shfl_xor(mx, 1, 32));
      mx = fmaxf(mx, __shfl_xor(mx, 2, 32));
      mx = fmaxf(mx, __shfl_xor(mx, 4, 32));
      mx = fmaxf(mx, __shfl_xor(mx, 8, 32));
      const float mnew = fmaxf(mrow[r], mx);
      const float alpha = __expf(mrow[r] - mnew);
      const float p0 = __expf(Sc[0][r] - mnew);
      const float p1 = __expf(Sc[1][r] - mnew);
      Sc[0][r] = p0; Sc[1][r] = p1;
      float ps = p0 + p1;
      ps += __shfl_xor(ps, 1, 32);
      ps += __shfl_xor(ps, 2, 32);
      ps += __shfl_xor(ps, 4, 32);
      ps += __shfl_xor(ps, 8, 32);
      lrow[r] = lrow[r] * alpha + ps;
      mrow[r] = mnew;
#pragma unroll
      for (int t = 0; t < 4; t++) O[t][r] *= alpha;
    }

    // C-layout -> A-layout for P via per-wave LDS (same-wave DS ops in-order).
    __bf16* P = &Pscr[wave][0];
#pragma unroll
    for (int nt = 0; nt < 2; nt++)
#pragma unroll
      for (int r = 0; r < 8; r++)
        P[(r + 8 * hf) * VSTRIDE + nt * 16 + lh] = (__bf16)Sc[nt][r];

    const __bf16* pp = P + lh * VSTRIDE + 8 * hf;
    v16bf Pf = bf_combine(*(const v8bf*)pp, *(const v8bf*)(pp + 16));

    // O(16x64) += P(16x32) @ V(32x64); V B-fragments from transposed LDS tile.
#pragma unroll
    for (int t = 0; t < 4; t++) {
      const __bf16* vb = &Vt[(t * 16 + lh) * VSTRIDE + 8 * hf];
      v16bf Vf = bf_combine(*(const v8bf*)vb, *(const v8bf*)(vb + 16));
      O[t] = wmma_bf16(Pf, Vf, O[t]);
    }
  }

  // Normalize and scatter context as bf16 in (B,S,D) layout.
#pragma unroll
  for (int r = 0; r < 8; r++) {
    const float inv = (lrow[r] > 0.0f) ? 1.0f / lrow[r] : 0.0f;
    const int s = qrow0 + r + 8 * hf;
    const size_t rowoff = ((size_t)b * S_ + s) * D_ + h * DH_;
#pragma unroll
    for (int t = 0; t < 4; t++)
      ctx[rowoff + t * 16 + lh] = (__bf16)(O[t][r] * inv);
  }
}

// ---------------------------------------------------------------------------
// Kernel 4: output projection.  ctx(bf16) @ Wo^T(bf16) + bo -> fp32 out.
// Same 32x64 wave tiling as the QKV kernel.
// ---------------------------------------------------------------------------
__global__ void __launch_bounds__(256)
out_proj_kernel(const __bf16* __restrict__ Xb, const __bf16* __restrict__ Wt,
                const float* __restrict__ bias, float* __restrict__ out) {
  const int lane = threadIdx.x & 31;
  const int wave = threadIdx.x >> 5;
  const int hf = lane >> 4, lh = lane & 15;
  const int wm = wave & 3, wn = wave >> 2;
  const int mbase = blockIdx.x * 128 + wm * 32;
  const int nbase = blockIdx.y * 128 + wn * 64;

  v8f acc[2][4];
#pragma unroll
  for (int mt = 0; mt < 2; mt++)
#pragma unroll
    for (int nt = 0; nt < 4; nt++)
#pragma unroll
      for (int i = 0; i < 8; i++) acc[mt][nt][i] = 0.0f;

  for (int k0 = 0; k0 < D_; k0 += 32) {
    v16bf A0, A1;
    {
      const __bf16* p = Xb + (size_t)(mbase + lh) * D_ + k0 + 8 * hf;
      A0 = bf_combine(*(const v8bf*)p, *(const v8bf*)(p + 16));
      const __bf16* q = Xb + (size_t)(mbase + 16 + lh) * D_ + k0 + 8 * hf;
      A1 = bf_combine(*(const v8bf*)q, *(const v8bf*)(q + 16));
    }
#pragma unroll
    for (int nt = 0; nt < 4; nt++) {
      const __bf16* p = Wt + (size_t)(nbase + nt * 16 + lh) * D_ + k0 + 8 * hf;
      v16bf Bf = bf_combine(*(const v8bf*)p, *(const v8bf*)(p + 16));
      acc[0][nt] = wmma_bf16(A0, Bf, acc[0][nt]);
      acc[1][nt] = wmma_bf16(A1, Bf, acc[1][nt]);
    }
  }

#pragma unroll
  for (int nt = 0; nt < 4; nt++) {
    const int col = nbase + nt * 16 + lh;
    const float bcol = bias[col];
#pragma unroll
    for (int mt = 0; mt < 2; mt++) {
#pragma unroll
      for (int r = 0; r < 8; r++) {
        const int row = mbase + mt * 16 + r + 8 * hf;
        out[(size_t)row * D_ + col] = acc[mt][nt][r] + bcol;
      }
    }
  }
}

// ---------------------------------------------------------------------------
// Workspace layout (bf16 elements):
//   qp/kp/vp/ctx: 4 x B*S*D  (4 x 16.78 MB)
//   Wt:           4 x D*D    (8.39 MB)        total ~75.5 MB
// ---------------------------------------------------------------------------
extern "C" void kernel_launch(void* const* d_in, const int* in_sizes, int n_in,
                              void* d_out, int out_size, void* d_ws, size_t ws_size,
                              hipStream_t stream) {
  (void)in_sizes; (void)n_in; (void)out_size; (void)ws_size;
  const float* query = (const float*)d_in[0];
  const float* key_  = (const float*)d_in[1];
  const float* value = (const float*)d_in[2];
  const float* mask  = (const float*)d_in[3];
  const float* Wq = (const float*)d_in[4];
  const float* bq = (const float*)d_in[5];
  const float* Wk = (const float*)d_in[6];
  const float* bk = (const float*)d_in[7];
  const float* Wv = (const float*)d_in[8];
  const float* bv = (const float*)d_in[9];
  const float* Wo = (const float*)d_in[10];
  const float* bo = (const float*)d_in[11];
  float* out = (float*)d_out;

  const size_t BSD = (size_t)B_ * S_ * D_;
  __bf16* qp    = (__bf16*)d_ws;
  __bf16* kp    = qp + BSD;
  __bf16* vp    = kp + BSD;
  __bf16* ctx   = vp + BSD;
  __bf16* WtAll = ctx + BSD;

  wconv_kernel<<<dim3((D_ * D_) / 256, 4), 256, 0, stream>>>(Wq, Wk, Wv, Wo, WtAll);
  qkv_proj_kernel<<<dim3((B_ * S_) / 128, D_ / 128, 3), 256, 0, stream>>>(
      query, key_, value, WtAll, bq, bk, bv, qp, kp, vp);
  flash_attn_kernel<<<dim3(S_ / 128, B_ * H_), 256, 0, stream>>>(qp, kp, vp, mask, ctx);
  out_proj_kernel<<<dim3((B_ * S_) / 128, D_ / 128), 256, 0, stream>>>(
      ctx, WtAll + (size_t)3 * D_ * D_, bo, out);
}
